// MultimodalDiffusionTransformer_58609123722032
// MI455X (gfx1250) — matrix operand order repaired and checked by reading, not compiled, not run on hardware
//
#include <hip/hip_runtime.h>
#include <hip/hip_bf16.h>
#include <math.h>

// Problem constants (match reference)
#define HH    512
#define BB    256
#define TT    1000
#define CLIPV 1000.0f
#define NBLK  64            // persistent grid: 64 blocks x 256 threads = 512 waves

typedef _Float16 v16h __attribute__((ext_vector_type(16)));
typedef _Float16 v8h  __attribute__((ext_vector_type(8)));
typedef float    v8f  __attribute__((ext_vector_type(8)));

// ---------------------------------------------------------------------------
// Coefficients: betas, cumprod(alpha), reversed-order scan coefficients.
// ---------------------------------------------------------------------------
__global__ void coeff_kernel(float* __restrict__ ca, float* __restrict__ cb,
                             float* __restrict__ cde) {
    if (threadIdx.x != 0 || blockIdx.x != 0) return;
    float ac = 1.0f;
    for (int i = 0; i < TT; ++i) {
        float beta    = 0.0001f + (0.02f - 0.0001f) * (float)i / (float)(TT - 1);
        float ac_prev = ac;                  // cumprod up to i-1 (1.0 at i=0)
        ac *= (1.0f - beta);
        int s = TT - 1 - i;                  // scan visits reversed order
        ca[s]  = sqrtf(ac);
        cb[s]  = sqrtf(1.0f - ac);
        cde[s] = sqrtf(1.0f - ac_prev) + sqrtf(beta);
    }
}

// ---------------------------------------------------------------------------
// te_act[t,j] = silu(t * tw1[j] + tb1[j])                (1000 x 512)
// ---------------------------------------------------------------------------
__global__ void te_act_kernel(const float* __restrict__ tw1,
                              const float* __restrict__ tb1,
                              float* __restrict__ te) {
    int idx = blockIdx.x * blockDim.x + threadIdx.x;
    if (idx >= TT * HH) return;
    int t = idx / HH, j = idx - t * HH;
    float u = (float)t * tw1[j] + tb1[j];
    te[idx] = u / (1.0f + __expf(-u));
}

// ---------------------------------------------------------------------------
// Plain f32 GEMM  C[M,N] = A[M,K] @ B[K,N] + bias[N], optional relu.
// One-time precompute / heads only (~1 GFLOP total) — not the hot path.
// ---------------------------------------------------------------------------
__global__ void gemm_f32_kernel(const float* __restrict__ A,
                                const float* __restrict__ Bm,
                                const float* __restrict__ bias,
                                float* __restrict__ C,
                                int M, int N, int K, int relu) {
    int idx = blockIdx.x * blockDim.x + threadIdx.x;
    if (idx >= M * N) return;
    int i = idx / N, j = idx - i * N;
    float acc = bias[j];
    const float* a = A + (size_t)i * K;
    for (int k = 0; k < K; ++k) acc = fmaf(a[k], Bm[(size_t)k * N + j], acc);
    if (relu) acc = acc > 0.0f ? acc : 0.0f;
    C[idx] = acc;
}

// ---------------------------------------------------------------------------
// Pack a row-major f32 512x512 weight (B[k][n]) into the f16 WMMA B-fragment
// layout Wp[((nt*16 + kc)*32 + lane)*16 + e]:
//   lanes 0-15  hold K = kc*32 + {0..7, 16..23},  N = lane
//   lanes 16-31 hold K = kc*32 + {8..15, 24..31}, N = lane-16
// ---------------------------------------------------------------------------
__global__ void pack_w_kernel(const float* __restrict__ W, _Float16* __restrict__ Wp) {
    int p = blockIdx.x * blockDim.x + threadIdx.x;    // 32*16*32*16 = 262144
    if (p >= (HH / 16) * (HH / 32) * 32 * 16) return;
    int e    = p & 15;
    int lane = (p >> 4) & 31;
    int kc   = (p >> 9) & 15;
    int nt   = p >> 13;
    int hi   = lane >> 4;
    int n    = nt * 16 + (lane & 15);
    int k    = kc * 32 + 8 * hi + e + ((e >> 3) << 3);
    Wp[p] = (_Float16)W[(size_t)k * HH + n];
}

// ---------------------------------------------------------------------------
// x init (f32 master + f16 mirror) and grid-barrier state init.
// ---------------------------------------------------------------------------
__global__ void init_x_kernel(const float* __restrict__ x0,
                              float* __restrict__ x32, _Float16* __restrict__ xh,
                              unsigned* __restrict__ bar) {
    int idx = blockIdx.x * blockDim.x + threadIdx.x;
    if (idx < 2) bar[idx] = 0u;              // [0]=arrive counter, [1]=generation
    if (idx >= BB * HH) return;
    float v = x0[idx];
    x32[idx] = v;
    xh[idx]  = (_Float16)v;
}

// ---------------------------------------------------------------------------
// A-fragment loader (row-major f16): lane (m=lane&15, hi=lane>>4) holds
// K = {8*hi..8*hi+7, 16+8*hi..+7} -> two contiguous 16B loads per k-chunk.
// ---------------------------------------------------------------------------
__device__ __forceinline__ v16h load_a_frag(const _Float16* __restrict__ arow,
                                            int kb, int hi) {
    v8h lo = *(const v8h*)(arow + kb + 8 * hi);
    v8h hh = *(const v8h*)(arow + kb + 16 + 8 * hi);
    return __builtin_shufflevector(lo, hh, 0, 1, 2, 3, 4, 5, 6, 7,
                                           8, 9, 10, 11, 12, 13, 14, 15);
}

// ---------------------------------------------------------------------------
// Device-wide generation barrier (persistent kernel, NBLK blocks co-resident).
// target is the (deterministic) generation this barrier completes.
// ---------------------------------------------------------------------------
__device__ __forceinline__ void grid_barrier(unsigned* __restrict__ bar,
                                             unsigned target) {
    __threadfence();                          // release my global stores (agent)
    __syncthreads();
    if (threadIdx.x == 0) {
        if (__hip_atomic_fetch_add(&bar[0], 1u, __ATOMIC_ACQ_REL,
                                   __HIP_MEMORY_SCOPE_AGENT) == (unsigned)(NBLK - 1)) {
            __hip_atomic_store(&bar[0], 0u, __ATOMIC_RELAXED, __HIP_MEMORY_SCOPE_AGENT);
            __hip_atomic_store(&bar[1], target, __ATOMIC_RELEASE, __HIP_MEMORY_SCOPE_AGENT);
        } else {
            while (__hip_atomic_load(&bar[1], __ATOMIC_ACQUIRE,
                                     __HIP_MEMORY_SCOPE_AGENT) < target) {
                __builtin_amdgcn_s_sleep(1);
            }
        }
    }
    __syncthreads();
    __threadfence();                          // acquire: drop stale WGP$ lines
}

// ---------------------------------------------------------------------------
// CDNA5 async global->LDS copy (bypasses VGPRs, tracked by ASYNCcnt).
// ---------------------------------------------------------------------------
__device__ __forceinline__ void async_copy_b128(unsigned lds_addr, const void* gptr) {
    asm volatile("global_load_async_to_lds_b128 %0, %1, off"
                 :: "v"(lds_addr), "v"(gptr) : "memory");
}
__device__ __forceinline__ void wait_asynccnt0() {
    asm volatile("s_wait_asynccnt 0x0" ::: "memory");
}

// ---------------------------------------------------------------------------
// Persistent scan kernel: all 1000 steps in ONE launch.
//   64 blocks x 256 thr; wave w: nl=w&1, ml=w>>1... mapping below.
//   Block owns 2 n-tiles; both weight B-panels for those n-tiles (4 x 16 KB =
//   64 KB) are async-copied to LDS once and reused for all 1000 steps.
//   Per step: GEMM1 (h1 = relu(x@w1 + tep)) | grid barrier |
//             GEMM2 (eps = h1@w2 + nb2) fused with x update | grid barrier.
// ---------------------------------------------------------------------------
__global__ void __launch_bounds__(256)
persistent_scan_kernel(const _Float16* __restrict__ w1p,
                       const _Float16* __restrict__ w2p,
                       const float* __restrict__ te_proj,
                       const float* __restrict__ nb2,
                       const float* __restrict__ ca,
                       const float* __restrict__ cb,
                       const float* __restrict__ cde,
                       float* __restrict__ x32,
                       _Float16* __restrict__ xh,
                       _Float16* __restrict__ h1h,
                       unsigned* __restrict__ bar) {
    __shared__ _Float16 smem[4][16 * 32 * 16];   // [w1(nl0),w1(nl1),w2(nl0),w2(nl1)]

    const int tid  = threadIdx.x;
    const int lane = tid & 31;
    const int wave = tid >> 5;
    const int hi   = lane >> 4;
    const int ln   = lane & 15;

    const int nt_blk = blockIdx.x & 15;          // 16 n-tile pairs
    const int mt_blk = blockIdx.x >> 4;          // 4 m-tile quads
    const int nl = wave & 1;                     // n-tile within pair
    const int ml = wave >> 1;                    // m-tile within quad
    const int nt = nt_blk * 2 + nl;
    const int mt = mt_blk * 4 + ml;
    const int ng = nt * 16 + ln;                 // global n for this lane

    // ---- One-time async fill of the block's weight panels into LDS ----
    {
        const unsigned lds_base = (unsigned)(size_t)&smem[0][0];
        for (int p = 0; p < 4; ++p) {            // panel: matrix = p>>1, nl = p&1
            const _Float16* src = ((p >> 1) ? w2p : w1p)
                                  + (size_t)(nt_blk * 2 + (p & 1)) * 8192;
#pragma unroll
            for (int i = 0; i < 4; ++i) {        // 4 x 16B per thread per panel
                unsigned boff = (unsigned)(tid * 64 + i * 16);
                async_copy_b128(lds_base + (unsigned)p * 16384u + boff,
                                (const char*)src + boff);
            }
        }
        wait_asynccnt0();
        __syncthreads();
    }

    const _Float16* bp1 = &smem[nl][0];          // GEMM1 B panel (this wave's nt)
    const _Float16* bp2 = &smem[2 + nl][0];      // GEMM2 B panel

    // ---- 1000 sequential diffusion steps ----
    for (int s = 0; s < TT; ++s) {
        // GEMM1: h1 = relu(x @ nw1x + te_proj[T-1-s])
        {
            const _Float16* arow = xh + (size_t)(mt * 16 + ln) * HH;
            v8f c = {};
#pragma unroll
            for (int kc = 0; kc < HH / 32; ++kc) {
                v16h a = load_a_frag(arow, kc * 32, hi);
                v16h b = *(const v16h*)(bp1 + (kc * 32 + lane) * 16);
                c = __builtin_amdgcn_wmma_f32_16x16x32_f16(false, a, false, b,
                                                           (short)0, c, false, false);
            }
            const float tep = te_proj[(size_t)(TT - 1 - s) * HH + ng];
#pragma unroll
            for (int r = 0; r < 8; ++r) {        // D: lane(n=ln), vgpr r -> m=r+8*hi
                float v = c[r] + tep;
                v = v > 0.0f ? v : 0.0f;
                h1h[(size_t)(mt * 16 + r + 8 * hi) * HH + ng] = (_Float16)v;
            }
        }
        grid_barrier(bar, (unsigned)(2 * s + 1));

        // GEMM2 + fused update: eps = h1@nw2 + nb2; x = clip((x-cb*eps)/ca + cde*eps)
        {
            const _Float16* arow = h1h + (size_t)(mt * 16 + ln) * HH;
            v8f c = {};
#pragma unroll
            for (int kc = 0; kc < HH / 32; ++kc) {
                v16h a = load_a_frag(arow, kc * 32, hi);
                v16h b = *(const v16h*)(bp2 + (kc * 32 + lane) * 16);
                c = __builtin_amdgcn_wmma_f32_16x16x32_f16(false, a, false, b,
                                                           (short)0, c, false, false);
            }
            const float bias = nb2[ng];
            const float A = ca[s], Bc = cb[s], D = cde[s];
#pragma unroll
            for (int r = 0; r < 8; ++r) {
                float  eps = c[r] + bias;
                size_t idx = (size_t)(mt * 16 + r + 8 * hi) * HH + ng;
                float  xv  = x32[idx];
                float  xn  = (xv - Bc * eps) / A + D * eps;
                xn = xn >  CLIPV ?  CLIPV : xn;
                xn = xn < -CLIPV ? -CLIPV : xn;
                x32[idx] = xn;
                xh[idx]  = (_Float16)xn;
            }
        }
        grid_barrier(bar, (unsigned)(2 * s + 2));
    }
}

// ---------------------------------------------------------------------------
// Output heads: out[b,0]=dep, out[b,1:3]=emo, out[b,3:515]=x.
// ---------------------------------------------------------------------------
__global__ void heads_kernel(const float* __restrict__ hd,
                             const float* __restrict__ he,
                             const float* __restrict__ dw2,
                             const float* __restrict__ db2,
                             const float* __restrict__ ew2,
                             const float* __restrict__ eb2,
                             const float* __restrict__ x32,
                             float* __restrict__ out) {
    const int b   = blockIdx.x;
    const int tid = threadIdx.x;
    for (int j = tid; j < HH; j += blockDim.x)
        out[(size_t)b * 515 + 3 + j] = x32[(size_t)b * HH + j];
    if (tid == 0) {
        float acc = db2[0];
        for (int j = 0; j < HH / 2; ++j) acc = fmaf(hd[b * (HH / 2) + j], dw2[j], acc);
        out[(size_t)b * 515] = acc;
    } else if (tid == 1 || tid == 2) {
        int ch = tid - 1;
        float acc = eb2[ch];
        for (int j = 0; j < HH / 2; ++j) acc = fmaf(he[b * (HH / 2) + j], ew2[j * 2 + ch], acc);
        out[(size_t)b * 515 + 1 + ch] = acc;
    }
}

// ---------------------------------------------------------------------------
// Host-side launch.
// ---------------------------------------------------------------------------
extern "C" void kernel_launch(void* const* d_in, const int* in_sizes, int n_in,
                              void* d_out, int out_size, void* d_ws, size_t ws_size,
                              hipStream_t stream) {
    const float* x0  = (const float*)d_in[0];
    const float* tw1 = (const float*)d_in[1];
    const float* tb1 = (const float*)d_in[2];
    const float* tw2 = (const float*)d_in[3];
    const float* tb2 = (const float*)d_in[4];
    const float* nw1 = (const float*)d_in[5];   // (2H,H): rows [0:H)=nw1x, [H:2H)=proj
    const float* nb1 = (const float*)d_in[6];
    const float* nw2 = (const float*)d_in[7];
    const float* nb2 = (const float*)d_in[8];
    const float* dw1 = (const float*)d_in[9];
    const float* db1 = (const float*)d_in[10];
    const float* dw2 = (const float*)d_in[11];
    const float* db2 = (const float*)d_in[12];
    const float* ew1 = (const float*)d_in[13];
    const float* eb1 = (const float*)d_in[14];
    const float* ew2 = (const float*)d_in[15];
    const float* eb2 = (const float*)d_in[16];
    float* out = (float*)d_out;

    uint8_t* wsb = (uint8_t*)d_ws;
    size_t off = 0;
    auto carve = [&](size_t bytes) -> void* {
        void* p = (void*)(wsb + off);
        off += (bytes + 255) & ~(size_t)255;
        return p;
    };
    float*     ca      = (float*)carve(TT * sizeof(float));
    float*     cbv     = (float*)carve(TT * sizeof(float));
    float*     cde     = (float*)carve(TT * sizeof(float));
    unsigned*  bar     = (unsigned*)carve(2 * sizeof(unsigned));
    float*     te_act  = (float*)carve((size_t)TT * HH * sizeof(float));
    float*     te_full = (float*)carve((size_t)TT * HH * sizeof(float));
    float*     te_proj = (float*)carve((size_t)TT * HH * sizeof(float));
    float*     x32     = (float*)carve((size_t)BB * HH * sizeof(float));
    float*     hd      = (float*)carve((size_t)BB * (HH / 2) * sizeof(float));
    float*     he      = (float*)carve((size_t)BB * (HH / 2) * sizeof(float));
    _Float16*  xh      = (_Float16*)carve((size_t)BB * HH * sizeof(_Float16));
    _Float16*  h1h     = (_Float16*)carve((size_t)BB * HH * sizeof(_Float16));
    _Float16*  w1p     = (_Float16*)carve((size_t)HH * HH * sizeof(_Float16));
    _Float16*  w2p     = (_Float16*)carve((size_t)HH * HH * sizeof(_Float16));
    (void)ws_size; (void)in_sizes; (void)n_in; (void)out_size;

    // --- One-time precompute ---------------------------------------------
    coeff_kernel<<<1, 32, 0, stream>>>(ca, cbv, cde);
    te_act_kernel<<<(TT * HH + 255) / 256, 256, 0, stream>>>(tw1, tb1, te_act);
    gemm_f32_kernel<<<(TT * HH + 255) / 256, 256, 0, stream>>>(
        te_act, tw2, tb2, te_full, TT, HH, HH, 0);
    gemm_f32_kernel<<<(TT * HH + 255) / 256, 256, 0, stream>>>(
        te_full, nw1 + (size_t)HH * HH, nb1, te_proj, TT, HH, HH, 0);
    pack_w_kernel<<<(HH * HH + 255) / 256, 256, 0, stream>>>(nw1, w1p);  // nw1[:H]
    pack_w_kernel<<<(HH * HH + 255) / 256, 256, 0, stream>>>(nw2, w2p);
    init_x_kernel<<<(BB * HH + 255) / 256, 256, 0, stream>>>(x0, x32, xh, bar);

    // --- Entire 1000-step scan in a single persistent launch --------------
    persistent_scan_kernel<<<NBLK, 256, 0, stream>>>(
        w1p, w2p, te_proj, nb2, ca, cbv, cde, x32, xh, h1h, bar);

    // --- Output heads ------------------------------------------------------
    gemm_f32_kernel<<<(BB * (HH / 2) + 255) / 256, 256, 0, stream>>>(
        x32, dw1, db1, hd, BB, HH / 2, HH, 1);
    gemm_f32_kernel<<<(BB * (HH / 2) + 255) / 256, 256, 0, stream>>>(
        x32, ew1, eb1, he, BB, HH / 2, HH, 1);
    heads_kernel<<<BB, 256, 0, stream>>>(hd, he, dw2, db2, ew2, eb2, x32, out);
}
